// ConditionalFeedForward_83468394430808
// MI455X (gfx1250) — matrix operand order repaired and checked by reading, not compiled, not run on hardware
//
#include <hip/hip_runtime.h>

// Problem constants (from reference): T=2048, A=2, E=8, D=2048, INTER=4096
#define AEXP   2
#define NEXP   8
#define DIM    2048
#define INTERD 4096
#define NASSIGN 4096

typedef __attribute__((ext_vector_type(16))) __bf16 v16bf;
typedef __attribute__((ext_vector_type(8)))  float  v8f;
typedef int v4i_vs __attribute__((vector_size(16)));   // matches builtin pointee type

constexpr int BM      = 128;  // token-tile (rows)
constexpr int BN      = 64;   // output-feature tile
constexpr int BK      = 32;   // K step (bf16 WMMA K)
constexpr int LDK     = 40;   // padded LDS row stride (bf16) -> conflict-free frag reads
constexpr int MSTRIPE = 4;    // m-tile stripes per (expert, n-tile)

// ---- gfx1250 async global->LDS (guarded) ---------------------------------
#if defined(__has_builtin)
#if __has_builtin(__builtin_amdgcn_global_load_async_to_lds_b128) && \
    __has_builtin(__builtin_amdgcn_s_wait_asynccnt)
#define HAVE_ASYNC_LDS 1
#endif
#endif
#ifndef HAVE_ASYNC_LDS
#define HAVE_ASYNC_LDS 0
#endif

#if HAVE_ASYNC_LDS
__device__ __forceinline__ void async_cp16(const unsigned short* g, unsigned short* l) {
  // as3 pointer value = low 32 bits of generic LDS address (ISA 10.2 aperture rule)
  __builtin_amdgcn_global_load_async_to_lds_b128(
      (__attribute__((address_space(1))) v4i_vs*)(uintptr_t)g,
      (__attribute__((address_space(3))) v4i_vs*)(unsigned)(uintptr_t)l,
      0, 0);
}
__device__ __forceinline__ void async_wait0() {
  __builtin_amdgcn_s_wait_asynccnt(0);
}
#endif

// ---- helpers -------------------------------------------------------------

// scalar fp32 -> bf16, RNE (epilogue only)
__device__ __forceinline__ unsigned short f2bf(float f) {
  union { float f; unsigned u; } v; v.f = f;
  unsigned u = v.u;
  return (unsigned short)((u + 0x7FFFu + ((u >> 16) & 1u)) >> 16);
}

// two fp32 -> packed 2xbf16: round-half-up add + single v_perm_b32
__device__ __forceinline__ unsigned pk2bf(float a, float b) {
  union { float f; unsigned u; } ua, ub; ua.f = a; ub.f = b;
  return __builtin_amdgcn_perm(ub.u + 0x8000u, ua.u + 0x8000u, 0x07060302u);
}

__device__ __forceinline__ uint4 cvt8r(float4 a, float4 b) {
  uint4 r;
  r.x = pk2bf(a.x, a.y);
  r.y = pk2bf(a.z, a.w);
  r.z = pk2bf(b.x, b.y);
  r.w = pk2bf(b.z, b.w);
  return r;
}

__device__ __forceinline__ v16bf frag_ld2(const unsigned short* p0,
                                          const unsigned short* p1) {
  union { uint4 q[2]; v16bf v; } u;
  u.q[0] = *(const uint4*)p0;
  u.q[1] = *(const uint4*)p1;
  return u.v;
}

__device__ __forceinline__ v8f wmma_bf16(v16bf a, v16bf b, v8f c) {
  return __builtin_amdgcn_wmma_f32_16x16x32_bf16(false, a, false, b, (short)0, c,
                                                 false, false);
}

// ---- kernel 0: group (t,a) assignments by expert -------------------------

__global__ __launch_bounds__(256) void moe_group_kernel(
    const int* __restrict__ idx, int* __restrict__ offsets,
    int* __restrict__ toklist) {
  __shared__ int cnt[NEXP], off[NEXP], cur[NEXP];
  const int tid = threadIdx.x;
  if (tid < NEXP) { cnt[tid] = 0; cur[tid] = 0; }
  __syncthreads();
  for (int i = tid; i < NASSIGN; i += 256) atomicAdd(&cnt[idx[i]], 1);
  __syncthreads();
  if (tid == 0) {
    int run = 0;
    for (int e = 0; e < NEXP; ++e) { off[e] = run; offsets[e] = run; run += cnt[e]; }
    offsets[NEXP] = run;
  }
  __syncthreads();
  for (int i = tid; i < NASSIGN; i += 256) {
    int e = idx[i];
    int p = off[e] + atomicAdd(&cur[e], 1);
    toklist[p] = i;
  }
}

// ---- kernel 1: H = silu(Xg W1^T) * (Xg W3^T), stored bf16 ---------------
// grid (INTER/BN, MSTRIPE, E), 256 thr = 8 waves (4 in M x 2 in N), wave tile
// 32x32 per matrix, double-buffered LDS, depth-2 register pipeline.

__global__ __launch_bounds__(256) void moe_h_kernel(
    const float* __restrict__ x, const float* __restrict__ w1,
    const float* __restrict__ w3, const int* __restrict__ offsets,
    const int* __restrict__ toklist, unsigned short* __restrict__ H) {
  const int e   = blockIdx.z;
  const int beg = offsets[e], end = offsets[e + 1];
  const int cnt = end - beg;
  const int n0  = blockIdx.x * BN;

  __shared__ unsigned short sA [2][BM * LDK];
  __shared__ unsigned short sB1[2][BN * LDK];
  __shared__ unsigned short sB3[2][BN * LDK];

  const int tid  = threadIdx.x;
  const int lane = tid & 31;
  const int wid  = tid >> 5;
  const int wm   = wid & 3;   // 4 waves in M, 32 rows each
  const int wn   = wid >> 2;  // 2 waves in N, 32 cols each

  const int rA = tid >> 1, cA = (tid & 1) * 16;  // A fill: 128 rows x 2 chunks
  const int rB = tid >> 2, cB = (tid & 3) * 8;   // B fill: 64 rows x 4 chunks

  const float* w1row = w1 + ((size_t)e * INTERD + (n0 + rB)) * DIM;
  const float* w3row = w3 + ((size_t)e * INTERD + (n0 + rB)) * DIM;

  const int ka0 = (lane >> 4) * 8;    // A frag: K {0-7,16-23} vs {8-15,24-31}
  const int kb0 = (lane >> 4) * 8;    // B frag: K {0-7,16-23} vs {8-15,24-31}
  const int hi  = lane >> 4, col = lane & 15;
  const int mlb = wm * 32 + (lane & 15);
  const int nlb = wn * 32 + (lane & 15);

  for (int mt = blockIdx.y; mt * BM < cnt; mt += MSTRIPE) {
    const int m0   = mt * BM;
    const int posA = beg + m0 + rA;
    const bool rowok = (posA < end);
    const float* xrow = rowok ? (x + (size_t)(toklist[posA] / AEXP) * DIM) : nullptr;

    v8f acc1[2][2] = {};
    v8f acc3[2][2] = {};

    float4 ax[4], b1s[2], b3s[2];
    auto fetch = [&](int kk) {
      if (rowok) {
        const float4* p = (const float4*)(xrow + kk + cA);
        ax[0] = p[0]; ax[1] = p[1]; ax[2] = p[2]; ax[3] = p[3];
      }
      { const float4* p = (const float4*)(w1row + kk + cB); b1s[0] = p[0]; b1s[1] = p[1]; }
      { const float4* p = (const float4*)(w3row + kk + cB); b3s[0] = p[0]; b3s[1] = p[1]; }
    };
    auto stage = [&](int p) {
      const uint4 z = make_uint4(0u, 0u, 0u, 0u);
      *(uint4*)&sA[p][rA * LDK + cA]     = rowok ? cvt8r(ax[0], ax[1]) : z;
      *(uint4*)&sA[p][rA * LDK + cA + 8] = rowok ? cvt8r(ax[2], ax[3]) : z;
      *(uint4*)&sB1[p][rB * LDK + cB] = cvt8r(b1s[0], b1s[1]);
      *(uint4*)&sB3[p][rB * LDK + cB] = cvt8r(b3s[0], b3s[1]);
    };

    fetch(0);
    stage(0);
    fetch(BK);
    __syncthreads();

    int p = 0;
    for (int kk = 0; kk < DIM; kk += BK) {
      v16bf afr[2];
#pragma unroll
      for (int mt2 = 0; mt2 < 2; ++mt2) {
        const int ml = mlb + mt2 * 16;
        afr[mt2] = frag_ld2(&sA[p][ml * LDK + ka0], &sA[p][ml * LDK + ka0 + 16]);
      }
#pragma unroll
      for (int nt = 0; nt < 2; ++nt) {
        const int nl = nlb + nt * 16;
        const v16bf b1 = frag_ld2(&sB1[p][nl * LDK + kb0], &sB1[p][nl * LDK + kb0 + 16]);
        const v16bf b3 = frag_ld2(&sB3[p][nl * LDK + kb0], &sB3[p][nl * LDK + kb0 + 16]);
#pragma unroll
        for (int mt2 = 0; mt2 < 2; ++mt2) {
          acc1[mt2][nt] = wmma_bf16(afr[mt2], b1, acc1[mt2][nt]);
          acc3[mt2][nt] = wmma_bf16(afr[mt2], b3, acc3[mt2][nt]);
        }
      }
      if (kk + BK < DIM) {
        stage(p ^ 1);
        if (kk + 2 * BK < DIM) fetch(kk + 2 * BK);
        if (kk + 3 * BK < DIM) {
          __builtin_prefetch(w1row + kk + 3 * BK + cB, 0, 1);
          __builtin_prefetch(w3row + kk + 3 * BK + cB, 0, 1);
        }
      }
      __syncthreads();
      p ^= 1;
    }

#pragma unroll
    for (int mt2 = 0; mt2 < 2; ++mt2) {
#pragma unroll
      for (int v = 0; v < 8; ++v) {
        const int p2 = beg + m0 + wm * 32 + mt2 * 16 + hi * 8 + v;
        if (p2 >= end) continue;
        const int aid = toklist[p2];
        const size_t base = (size_t)aid * INTERD + n0 + wn * 32 + col;
#pragma unroll
        for (int nt = 0; nt < 2; ++nt) {
          const float a = acc1[mt2][nt][v];
          const float b = acc3[mt2][nt][v];
          const float h = (a / (1.0f + __expf(-a))) * b;   // silu(a) * b
          H[base + nt * 16] = f2bf(h);
        }
      }
    }
  }
}

// ---- kernel 2: Out = H W2^T, fp32 scatter --------------------------------
// grid (D/BN, MSTRIPE, E). A-tile (bf16 H rows) staged via async global->LDS
// when available; w2 converted fp32->bf16 through registers.

__global__ __launch_bounds__(256) void moe_out_kernel(
    const unsigned short* __restrict__ H, const float* __restrict__ w2,
    const int* __restrict__ offsets, const int* __restrict__ toklist,
    float* __restrict__ out) {
  const int e   = blockIdx.z;
  const int beg = offsets[e], end = offsets[e + 1];
  const int cnt = end - beg;
  const int n0  = blockIdx.x * BN;

  __shared__ unsigned short sA[2][BM * LDK];
  __shared__ unsigned short sB[2][BN * LDK];

  const int tid  = threadIdx.x;
  const int lane = tid & 31;
  const int wid  = tid >> 5;
  const int wm   = wid & 3, wn = wid >> 2;

  const int rA = tid >> 1, cA = (tid & 1) * 16;
  const int rB = tid >> 2, cB = (tid & 3) * 8;

  const float* w2row = w2 + ((size_t)e * DIM + (n0 + rB)) * INTERD;

  const int ka0 = (lane >> 4) * 8;
  const int kb0 = (lane >> 4) * 8;
  const int hi  = lane >> 4, col = lane & 15;
  const int mlb = wm * 32 + (lane & 15);
  const int nlb = wn * 32 + (lane & 15);

  for (int mt = blockIdx.y; mt * BM < cnt; mt += MSTRIPE) {
    const int m0   = mt * BM;
    const int posA = beg + m0 + rA;
    const bool rowok = (posA < end);
    const unsigned short* hrow =
        rowok ? (H + (size_t)toklist[posA] * INTERD) : nullptr;

    v8f acc[2][2] = {};

    float4 fb0, fb1;
    auto fetchB = [&](int kk) {
      const float4* p = (const float4*)(w2row + kk + cB);
      fb0 = p[0]; fb1 = p[1];
    };
    auto stageB = [&](int p) {
      *(uint4*)&sB[p][rB * LDK + cB] = cvt8r(fb0, fb1);
    };

#if HAVE_ASYNC_LDS
    // async writes are EXEC-masked: pre-zero rows with no assignment once
    if (!rowok) {
      const uint4 z = make_uint4(0u, 0u, 0u, 0u);
      *(uint4*)&sA[0][rA * LDK + cA]     = z;
      *(uint4*)&sA[0][rA * LDK + cA + 8] = z;
      *(uint4*)&sA[1][rA * LDK + cA]     = z;
      *(uint4*)&sA[1][rA * LDK + cA + 8] = z;
    }
    auto stageA = [&](int p, int kk) {
      if (rowok) {
        async_cp16(hrow + kk + cA,     &sA[p][rA * LDK + cA]);
        async_cp16(hrow + kk + cA + 8, &sA[p][rA * LDK + cA + 8]);
      }
    };
    stageA(0, 0);
    fetchB(0);
    stageB(0);
    fetchB(BK);
    async_wait0();
    __syncthreads();
#else
    uint4 ha0 = make_uint4(0u, 0u, 0u, 0u), ha1 = ha0;
    auto fetchA = [&](int kk) {
      if (rowok) {
        const uint4* p = (const uint4*)(hrow + kk + cA);
        ha0 = p[0]; ha1 = p[1];
      }
    };
    auto stageA = [&](int p) {
      *(uint4*)&sA[p][rA * LDK + cA]     = ha0;
      *(uint4*)&sA[p][rA * LDK + cA + 8] = ha1;
    };
    fetchA(0);
    fetchB(0);
    stageA(0);
    stageB(0);
    fetchA(BK);
    fetchB(BK);
    __syncthreads();
#endif

    int p = 0;
    for (int kk = 0; kk < INTERD; kk += BK) {
      v16bf afr[2];
#pragma unroll
      for (int mt2 = 0; mt2 < 2; ++mt2) {
        const int ml = mlb + mt2 * 16;
        afr[mt2] = frag_ld2(&sA[p][ml * LDK + ka0], &sA[p][ml * LDK + ka0 + 16]);
      }
#pragma unroll
      for (int nt = 0; nt < 2; ++nt) {
        const int nl = nlb + nt * 16;
        const v16bf bf = frag_ld2(&sB[p][nl * LDK + kb0], &sB[p][nl * LDK + kb0 + 16]);
#pragma unroll
        for (int mt2 = 0; mt2 < 2; ++mt2)
          acc[mt2][nt] = wmma_bf16(afr[mt2], bf, acc[mt2][nt]);
      }
      if (kk + BK < INTERD) {
#if HAVE_ASYNC_LDS
        stageA(p ^ 1, kk + BK);
        stageB(p ^ 1);
        if (kk + 2 * BK < INTERD) fetchB(kk + 2 * BK);
#else
        stageA(p ^ 1);
        stageB(p ^ 1);
        if (kk + 2 * BK < INTERD) { fetchA(kk + 2 * BK); fetchB(kk + 2 * BK); }
#endif
        if (kk + 3 * BK < INTERD)
          __builtin_prefetch(w2row + kk + 3 * BK + cB, 0, 1);
      }
#if HAVE_ASYNC_LDS
      async_wait0();
#endif
      __syncthreads();
      p ^= 1;
    }

#pragma unroll
    for (int mt2 = 0; mt2 < 2; ++mt2) {
#pragma unroll
      for (int v = 0; v < 8; ++v) {
        const int p2 = beg + m0 + wm * 32 + mt2 * 16 + hi * 8 + v;
        if (p2 >= end) continue;
        const int aid = toklist[p2];
        const size_t base = (size_t)aid * DIM + n0 + wn * 32 + col;
#pragma unroll
        for (int nt = 0; nt < 2; ++nt) out[base + nt * 16] = acc[mt2][nt][v];
      }
    }
  }
}

// ---- launcher ------------------------------------------------------------
// inputs: x f32, expert_indices i32, w1 f32, w2 f32, w3 f32
// workspace: [0,64) offsets | [64,16448) toklist | [16448, +32MB) H (bf16)

extern "C" void kernel_launch(void* const* d_in, const int* in_sizes, int n_in,
                              void* d_out, int out_size, void* d_ws, size_t ws_size,
                              hipStream_t stream) {
  const float* x  = (const float*)d_in[0];
  const int*   ix = (const int*)d_in[1];
  const float* w1 = (const float*)d_in[2];
  const float* w2 = (const float*)d_in[3];
  const float* w3 = (const float*)d_in[4];
  float* out = (float*)d_out;

  int* offsets = (int*)d_ws;
  int* toklist = offsets + 16;
  unsigned short* H = (unsigned short*)(toklist + NASSIGN);

  moe_group_kernel<<<1, 256, 0, stream>>>(ix, offsets, toklist);

  moe_h_kernel<<<dim3(INTERD / BN, MSTRIPE, NEXP), 256, 0, stream>>>(
      x, w1, w3, offsets, toklist, H);

  moe_out_kernel<<<dim3(DIM / BN, MSTRIPE, NEXP), 256, 0, stream>>>(
      H, w2, offsets, toklist, out);
}
